// MultiHeadAttention_28570122453151
// MI455X (gfx1250) — matrix operand order repaired and checked by reading, not compiled, not run on hardware
//
#include <hip/hip_runtime.h>
#include <cstdint>

// ---------------------------------------------------------------------------
// MI455X (gfx1250) fused MHA: all GEMMs via v_wmma_f32_16x16x32_f16 (wave32).
// LDS staging via gfx1250 async-to-LDS (ASYNCcnt) with double buffering so
// the async DMA of tile k+1 overlaps the WMMA work on tile k.
// ---------------------------------------------------------------------------

typedef __attribute__((ext_vector_type(16))) _Float16 v16h;
typedef __attribute__((ext_vector_type(8)))  _Float16 v8h;
typedef __attribute__((ext_vector_type(4)))  _Float16 v4h;
typedef __attribute__((ext_vector_type(8)))  float    v8f;

union V16 { v16h v; v8h h[2]; };

constexpr int NB = 4, SEQ = 2048, EMB = 1024, HEADS = 16, HD = 64;
constexpr int MROWS = NB * SEQ;  // 8192

// ---- gfx1250 async global->LDS copy (ASYNCcnt), with safe fallback --------
#if defined(__has_builtin)
#if __has_builtin(__builtin_amdgcn_global_load_async_to_lds_b128) && \
    __has_builtin(__builtin_amdgcn_s_wait_asynccnt)
#define USE_ASYNC_LDS 1
#endif
#endif

typedef int i32x4 __attribute__((ext_vector_type(4)));
typedef __attribute__((address_space(1))) i32x4 i32x4_g;   // global (AS1)
typedef __attribute__((address_space(3))) i32x4 i32x4_l;   // LDS (AS3)

__device__ __forceinline__ void copy16B(_Float16* dst_lds, const _Float16* src_gbl) {
#if defined(USE_ASYNC_LDS)
  // flat LDS address low 32 bits == LDS byte offset (ISA 10.2 aperture rules)
  __builtin_amdgcn_global_load_async_to_lds_b128(
      (i32x4_g*)(unsigned long long)src_gbl,
      (i32x4_l*)(unsigned int)(unsigned long long)dst_lds, 0, 0);
#else
  *(v8h*)dst_lds = *(const v8h*)src_gbl;
#endif
}

__device__ __forceinline__ void stage_fence() {
#if defined(USE_ASYNC_LDS)
  __builtin_amdgcn_s_wait_asynccnt(0);
#endif
  __syncthreads();
}

// ---- f32 -> f16 convert, 4-wide ------------------------------------------
__global__ void cvt_f32_f16(const float* __restrict__ s, _Float16* __restrict__ d, int n4) {
  int i = blockIdx.x * blockDim.x + threadIdx.x;
  if (i < n4) {
    const float4 f = ((const float4*)s)[i];
    v4h h = {(_Float16)f.x, (_Float16)f.y, (_Float16)f.z, (_Float16)f.w};
    ((v4h*)d)[i] = h;
  }
}

// ---- mask (int32)[S,T] -> transposed additive f16 bias [T,S], once --------
__global__ void mask_biasT(const int* __restrict__ m, _Float16* __restrict__ b, int n) {
  int i = blockIdx.x * blockDim.x + threadIdx.x;  // i = t*SEQ + s
  if (i < n) {
    const int tt = i >> 11, ss = i & (SEQ - 1);
    b[i] = m[ss * SEQ + tt] ? (_Float16)0.f : (_Float16)(-30000.f);
  }
}

// ---------------------------------------------------------------------------
// GEMM: C[M,N] = A[M,K] * W[N,K]^T + bias   (A,W f16; acc f32)
// Block: 256 thr = 8 waves, tile 128x128, wave tile 32x64, K-step 32,
// double-buffered async LDS staging.
// MODE 0: f16 out, head-split [n,h,s,d]   (Q, K)
// MODE 1: f16 out, head-split transposed [n,h,d,t]   (V^T)
// MODE 2: f32 out, row-major [M,N]        (final projection)
// ---------------------------------------------------------------------------
template <int MODE>
__global__ void __launch_bounds__(256)
gemm_wmma(const _Float16* __restrict__ A, const _Float16* __restrict__ W,
          const float* __restrict__ bias, void* __restrict__ OutP) {
  __shared__ _Float16 Al[2][128 * 40];  // stride 40 halves (80B, 16B-aligned)
  __shared__ _Float16 Bl[2][128 * 40];

  const int m0 = blockIdx.y * 128;
  const int n0 = blockIdx.x * 128;
  const int t = threadIdx.x;
  const int lane = t & 31;
  const int wv = t >> 5;
  const int wm = (wv >> 1) * 32;   // 4 wave-rows x 2 wave-cols
  const int wn = (wv & 1) * 64;
  const int arow = lane & 15;      // A frag: lane holds row, 16 contiguous K
  const int ak8 = (lane >> 4) * 16;
  const int bk16 = (lane >> 4) * 16;  // B frag: lane holds col, 16 contiguous K
  const int lrow = t >> 1;            // staging: 2 threads per 32-half row
  const int lkp = (t & 1) * 16;

  auto stage = [&](int buf, int k0) {
    copy16B(&Al[buf][lrow * 40 + lkp],     A + (size_t)(m0 + lrow) * EMB + k0 + lkp);
    copy16B(&Al[buf][lrow * 40 + lkp + 8], A + (size_t)(m0 + lrow) * EMB + k0 + lkp + 8);
    copy16B(&Bl[buf][lrow * 40 + lkp],     W + (size_t)(n0 + lrow) * EMB + k0 + lkp);
    copy16B(&Bl[buf][lrow * 40 + lkp + 8], W + (size_t)(n0 + lrow) * EMB + k0 + lkp + 8);
  };

  v8f acc[2][4] = {};
  int buf = 0;
  stage(0, 0);

  constexpr int NK = EMB / 32;
  for (int kt = 0; kt < NK; ++kt) {
    stage_fence();                           // tile kt resident in LDS
    if (kt + 1 < NK) stage(buf ^ 1, (kt + 1) * 32);  // async DMA overlaps WMMAs

    V16 a[2], b[4];
#pragma unroll
    for (int im = 0; im < 2; ++im) {
      const _Float16* p = &Al[buf][(wm + im * 16 + arow) * 40 + ak8];
      a[im].h[0] = *(const v8h*)p;
      a[im].h[1] = *(const v8h*)(p + 8);
    }
#pragma unroll
    for (int in = 0; in < 4; ++in) {
      const _Float16* p = &Bl[buf][(wn + in * 16 + (lane & 15)) * 40 + bk16];
      b[in].h[0] = *(const v8h*)p;
      b[in].h[1] = *(const v8h*)(p + 8);
    }
#pragma unroll
    for (int im = 0; im < 2; ++im)
#pragma unroll
      for (int in = 0; in < 4; ++in)
        acc[im][in] = __builtin_amdgcn_wmma_f32_16x16x32_f16(
            false, a[im].v, false, b[in].v, (short)0, acc[im][in], false, false);
    buf ^= 1;
  }

  // epilogue: C layout = vgpr r -> row r + 8*(lane>>4), col lane&15
#pragma unroll
  for (int im = 0; im < 2; ++im)
#pragma unroll
    for (int in = 0; in < 4; ++in)
#pragma unroll
      for (int r = 0; r < 8; ++r) {
        const int gm = m0 + wm + im * 16 + r + 8 * (lane >> 4);
        const int gn = n0 + wn + in * 16 + (lane & 15);
        const float val = acc[im][in][r] + bias[gn];
        if constexpr (MODE == 2) {
          ((float*)OutP)[(size_t)gm * EMB + gn] = val;
        } else {
          _Float16* o16 = (_Float16*)OutP;
          const int bb = gm >> 11, ss = gm & (SEQ - 1);
          const int hh = gn >> 6, dd = gn & (HD - 1);
          if constexpr (MODE == 0)
            o16[((size_t)(bb * HEADS + hh) * SEQ + ss) * HD + dd] = (_Float16)val;
          else
            o16[((size_t)(bb * HEADS + hh) * HD + dd) * SEQ + ss] = (_Float16)val;
        }
      }
}

// ---------------------------------------------------------------------------
// Flash attention per (batch*head). Block = 128 thr = 4 waves; wave owns 16
// Q rows. K / V^T / biasT tiles double-buffered in LDS: async DMA of tile
// t0+64 overlaps the 16 WMMAs + softmax of tile t0.
// ---------------------------------------------------------------------------
__global__ void __launch_bounds__(128)
attn_fa(const _Float16* __restrict__ Q, const _Float16* __restrict__ K,
        const _Float16* __restrict__ Vt, const _Float16* __restrict__ BiasT,
        _Float16* __restrict__ AO) {
  __shared__ _Float16 Kl[2][64 * 72];  // rows = t (64), cols = d (64)
  __shared__ _Float16 Vl[2][64 * 72];  // rows = d (64), cols = t (64)
  __shared__ _Float16 Bi[2][64 * 72];  // rows = t (64), cols = s-in-block (64)
  __shared__ _Float16 Pl[4][16 * 72];  // per-wave P tile

  const int nh = blockIdx.y;
  const int wv = threadIdx.x >> 5;
  const int lane = threadIdx.x & 31;
  const int s0b = blockIdx.x * 64;
  const int s0 = s0b + wv * 16;
  const int arow = lane & 15;
  const int ak8 = (lane >> 4) * 16;
  const int bk16 = (lane >> 4) * 16;
  const int myrow = wv * 16 + 8 * (lane >> 4);  // row within 64-row block (+r)
  const int srow = s0b + myrow;                 // row within SEQ (+r)
  _Float16* pl = &Pl[wv][0];
  const int lrow = threadIdx.x >> 1;            // staging: 2 thr per 64-half row
  const int lkp = (threadIdx.x & 1) * 32;

  auto stage = [&](int buf, int tt) {
    const _Float16* ks = K + ((size_t)nh * SEQ + tt + lrow) * HD + lkp;
    const _Float16* vs = Vt + ((size_t)nh * HD + lrow) * SEQ + tt + lkp;
    const _Float16* bs = BiasT + (size_t)(tt + lrow) * SEQ + s0b + lkp;
#pragma unroll
    for (int c = 0; c < 4; ++c) {
      copy16B(&Kl[buf][lrow * 72 + lkp + c * 8], ks + c * 8);
      copy16B(&Vl[buf][lrow * 72 + lkp + c * 8], vs + c * 8);
      copy16B(&Bi[buf][lrow * 72 + lkp + c * 8], bs + c * 8);
    }
  };

  // Q A-fragments (16 rows x 64 d) -> two 16x32 frags, kept resident
  V16 qa[2];
  {
    const _Float16* qb = Q + ((size_t)nh * SEQ + s0 + arow) * HD;
#pragma unroll
    for (int kk = 0; kk < 2; ++kk) {
      qa[kk].h[0] = *(const v8h*)(qb + kk * 32 + ak8);
      qa[kk].h[1] = *(const v8h*)(qb + kk * 32 + ak8 + 8);
    }
  }

  v8f o[4] = {};
  float mst[8], lst[8];
#pragma unroll
  for (int r = 0; r < 8; ++r) { mst[r] = -1e30f; lst[r] = 0.f; }

  int buf = 0;
  stage(0, 0);

  for (int t0 = 0; t0 < SEQ; t0 += 64) {
    stage_fence();                              // tile t0 resident in LDS
    if (t0 + 64 < SEQ) stage(buf ^ 1, t0 + 64); // async DMA overlaps compute

    // ---- scores S = Q K^T (16x64): 8 WMMAs, K frags from LDS ----
    v8f sc[4];
#pragma unroll
    for (int jn = 0; jn < 4; ++jn) {
      const _Float16* kp = &Kl[buf][(jn * 16 + (lane & 15)) * 72];
      V16 b0, b1;
      b0.h[0] = *(const v8h*)(kp + bk16);
      b0.h[1] = *(const v8h*)(kp + bk16 + 8);
      b1.h[0] = *(const v8h*)(kp + 32 + bk16);
      b1.h[1] = *(const v8h*)(kp + 32 + bk16 + 8);
      v8f c = {};
      c = __builtin_amdgcn_wmma_f32_16x16x32_f16(false, qa[0].v, false, b0.v, (short)0, c, false, false);
      c = __builtin_amdgcn_wmma_f32_16x16x32_f16(false, qa[1].v, false, b1.v, (short)0, c, false, false);
      sc[jn] = c;
    }
    // ---- scale 1/sqrt(64) + additive mask bias (transposed tile: the 8
    //      s-rows a lane owns are contiguous halves -> one b128 per jn) ----
#pragma unroll
    for (int jn = 0; jn < 4; ++jn) {
      const v8h bh = *(const v8h*)&Bi[buf][(jn * 16 + (lane & 15)) * 72 + myrow];
#pragma unroll
      for (int r = 0; r < 8; ++r)
        sc[jn][r] = sc[jn][r] * 0.125f + (float)bh[r];
    }
    // ---- online softmax: row max via xor-butterfly over the 16 lanes/row ----
#pragma unroll
    for (int r = 0; r < 8; ++r) {
      float v = fmaxf(fmaxf(sc[0][r], sc[1][r]), fmaxf(sc[2][r], sc[3][r]));
#pragma unroll
      for (int m = 1; m < 16; m <<= 1) v = fmaxf(v, __shfl_xor(v, m, 32));
      const float mnew = fmaxf(mst[r], v);
      const float alpha = exp2f((mst[r] - mnew) * 1.4426950408889634f);
      mst[r] = mnew;
      lst[r] *= alpha;
#pragma unroll
      for (int jd = 0; jd < 4; ++jd) o[jd][r] *= alpha;
    }
    // ---- P = exp(S - m), row sums, C-layout -> LDS (per-wave region) ----
    float rs[8] = {0.f, 0.f, 0.f, 0.f, 0.f, 0.f, 0.f, 0.f};
#pragma unroll
    for (int jn = 0; jn < 4; ++jn)
#pragma unroll
      for (int r = 0; r < 8; ++r) {
        const float p = exp2f((sc[jn][r] - mst[r]) * 1.4426950408889634f);
        rs[r] += p;
        pl[(r + 8 * (lane >> 4)) * 72 + jn * 16 + (lane & 15)] = (_Float16)p;
      }
#pragma unroll
    for (int r = 0; r < 8; ++r) {
      float v = rs[r];
#pragma unroll
      for (int m = 1; m < 16; m <<= 1) v += __shfl_xor(v, m, 32);
      lst[r] += v;
    }
    // ---- P A-frags from LDS (intra-wave: DS pipe is in-order, no barrier) ----
    V16 pa[2];
#pragma unroll
    for (int kk = 0; kk < 2; ++kk) {
      const _Float16* p = pl + arow * 72 + kk * 32 + ak8;
      pa[kk].h[0] = *(const v8h*)p;
      pa[kk].h[1] = *(const v8h*)(p + 8);
    }
    // ---- O += P V : 8 WMMAs, V frags from LDS ----
#pragma unroll
    for (int jd = 0; jd < 4; ++jd) {
      const _Float16* vp = &Vl[buf][(jd * 16 + (lane & 15)) * 72];
      V16 v0, v1;
      v0.h[0] = *(const v8h*)(vp + bk16);
      v0.h[1] = *(const v8h*)(vp + bk16 + 8);
      v1.h[0] = *(const v8h*)(vp + 32 + bk16);
      v1.h[1] = *(const v8h*)(vp + 32 + bk16 + 8);
      o[jd] = __builtin_amdgcn_wmma_f32_16x16x32_f16(false, pa[0].v, false, v0.v, (short)0, o[jd], false, false);
      o[jd] = __builtin_amdgcn_wmma_f32_16x16x32_f16(false, pa[1].v, false, v1.v, (short)0, o[jd], false, false);
    }
    buf ^= 1;
  }

  // ---- epilogue: O / l  -> f16 row-major [n*s, h*d] for final projection ----
  const int bb = nh >> 4, hh = nh & 15;
#pragma unroll
  for (int jd = 0; jd < 4; ++jd)
#pragma unroll
    for (int r = 0; r < 8; ++r) {
      const float val = o[jd][r] / lst[r];
      const int grow = bb * SEQ + srow + r;
      const int gcol = hh * HD + jd * 16 + (lane & 15);
      AO[(size_t)grow * EMB + gcol] = (_Float16)val;
    }
}

// ---------------------------------------------------------------------------
extern "C" void kernel_launch(void* const* d_in, const int* in_sizes, int n_in,
                              void* d_out, int out_size, void* d_ws, size_t ws_size,
                              hipStream_t stream) {
  (void)in_sizes; (void)n_in; (void)out_size; (void)ws_size;
  const float* q = (const float*)d_in[0];
  const float* k = (const float*)d_in[1];
  const float* v = (const float*)d_in[2];
  const int* mask = (const int*)d_in[3];
  const float* Wq = (const float*)d_in[4];
  const float* bq = (const float*)d_in[5];
  const float* Wk = (const float*)d_in[6];
  const float* bk = (const float*)d_in[7];
  const float* Wv = (const float*)d_in[8];
  const float* bv = (const float*)d_in[9];
  const float* Wo = (const float*)d_in[10];
  const float* bo = (const float*)d_in[11];

  const size_t NX = (size_t)MROWS * EMB;  // 8M elems
  const size_t NW = (size_t)EMB * EMB;    // 1M elems
  const size_t NM = (size_t)SEQ * SEQ;    // 4M elems
  _Float16* ws = (_Float16*)d_ws;
  _Float16* xq16 = ws;              // activations f16
  _Float16* xk16 = xq16 + NX;
  _Float16* xv16 = xk16 + NX;
  _Float16* wq16 = xv16 + NX;       // weights f16
  _Float16* wk16 = wq16 + NW;
  _Float16* wv16 = wk16 + NW;
  _Float16* wo16 = wv16 + NW;
  _Float16* Q16 = wo16 + NW;        // projected, head-split
  _Float16* K16 = Q16 + NX;
  _Float16* Vt16 = K16 + NX;        // V transposed per head [n,h,d,t]
  _Float16* AO16 = Vt16 + NX;       // attention output, row-major
  _Float16* BIT16 = AO16 + NX;      // f16 additive mask bias, transposed [T,S]
  // total: 7*NX + 4*NW + NM halves = ~128 MB of workspace

  auto cvt = [&](const float* s, _Float16* d, size_t n) {
    cvt_f32_f16<<<dim3((unsigned)(n / 4 / 256)), dim3(256), 0, stream>>>(s, d, (int)(n / 4));
  };
  cvt(q, xq16, NX);
  cvt(k, xk16, NX);
  cvt(v, xv16, NX);
  cvt(Wq, wq16, NW);
  cvt(Wk, wk16, NW);
  cvt(Wv, wv16, NW);
  cvt(Wo, wo16, NW);
  mask_biasT<<<dim3((unsigned)(NM / 256)), dim3(256), 0, stream>>>(mask, BIT16, (int)NM);

  const dim3 gg(EMB / 128, MROWS / 128);  // (8, 64)
  gemm_wmma<0><<<gg, 256, 0, stream>>>(xq16, wq16, bq, Q16);
  gemm_wmma<0><<<gg, 256, 0, stream>>>(xk16, wk16, bk, K16);
  gemm_wmma<1><<<gg, 256, 0, stream>>>(xv16, wv16, bv, Vt16);

  attn_fa<<<dim3(SEQ / 64, NB * HEADS), 128, 0, stream>>>(Q16, K16, Vt16, BIT16, AO16);

  gemm_wmma<2><<<gg, 256, 0, stream>>>(AO16, wo16, bo, d_out);
}
